// WindowedAttention_6674379178341
// MI455X (gfx1250) — matrix operand order repaired and checked by reading, not compiled, not run on hardware
//
#include <hip/hip_runtime.h>
#include <hip/hip_bf16.h>

typedef __bf16 bf16;
typedef bf16  bf16x16 __attribute__((ext_vector_type(16)));
typedef bf16  bf16x8  __attribute__((ext_vector_type(8)));
typedef bf16  bf16x2  __attribute__((ext_vector_type(2)));
typedef float floatx8 __attribute__((ext_vector_type(8)));

#define NUM_HEADS 16
#define HEAD_DIM  64
#define D_MODEL   1024
#define WINDOW    64
#define QK_SCALE  0.125f   // 8.0 / HEAD_DIM
#define LN_EPS    1e-5f

#define M_TOT  32768       // 4 * 8192 token rows
#define N_QKV  3072
#define K_DIM  1024

// ---------------- WMMA helpers (CDNA5 16x16x32 bf16, ISA 7.12.2) ----------------

__device__ __forceinline__ floatx8 wmma_bf16(bf16x16 a, bf16x16 b, floatx8 c) {
  return __builtin_amdgcn_wmma_f32_16x16x32_bf16(false, a, false, b, (short)0, c,
                                                 false, false);
}

// A fragment from row-major [M][K] LDS tile.
// row = row0 + lane%16; elem i -> K = k0 + h*16 + i.
// 32B contiguous -> 2x ds_load_b128.
__device__ __forceinline__ bf16x16 load_a(const bf16* lds, int row0, int k0, int ld,
                                          int lane) {
  int r = lane & 15, h = lane >> 4;
  const bf16* p = lds + (row0 + r) * ld + k0;
  bf16x16 a;
#pragma unroll
  for (int i = 0; i < 8; ++i) {
    a[i]     = p[h * 16 + i];
    a[i + 8] = p[h * 16 + 8 + i];
  }
  return a;
}

// B fragment from an LDS tile stored K-contiguous per column: X[n][k] == B[k][n].
// col = col0 + lane%16; elem i -> K = k0 + h*16 + i.  32B contiguous -> 2x b128.
__device__ __forceinline__ bf16x16 load_b_ct(const bf16* lds, int k0, int col0, int ld,
                                             int lane) {
  int r = lane & 15, h = lane >> 4;
  const bf16* p = lds + (col0 + r) * ld + k0 + h * 16;
  bf16x16 b;
#pragma unroll
  for (int i = 0; i < 16; ++i) b[i] = p[i];
  return b;
}

__device__ __forceinline__ bf16x2 pack_bf16(float x, float y) {
  bf16x2 r;
  r[0] = (bf16)x;   // fuses into v_cvt_pk_bf16_f32
  r[1] = (bf16)y;
  return r;
}

// CDNA5 async global->LDS copy (ASYNCcnt-tracked), 16 bytes per issue.
__device__ __forceinline__ void async_load_b128(unsigned int lds_off, const void* gaddr) {
  asm volatile("global_load_async_to_lds_b128 %0, %1, off"
               :: "v"(lds_off), "v"((unsigned long long)(size_t)gaddr)
               : "memory");
}
__device__ __forceinline__ void wait_asynccnt0() {
  asm volatile("s_wait_asynccnt 0x0" ::: "memory");
}

// ---------------- tiling ----------------
#define TM  128
#define TN  128
#define TK  64
#define TKP (TK + 8)   // 144B rows, keeps 16B alignment

// ---------------- GEMM 1: qkv = X(f32) @ W_qkv(f32) -> bf16 ----------------

__global__ __launch_bounds__(256) void gemm_qkv_kernel(const float* __restrict__ A,
                                                       const float* __restrict__ B,
                                                       bf16* __restrict__ C) {
  __shared__ bf16 As[TM][TKP];   // row-major [M][K]
  __shared__ bf16 Bt[TN][TKP];   // transposed: Bt[n][k] = B[k][n]
  const int bm = blockIdx.x * TM;
  const int bn = blockIdx.y * TN;
  const int tid = threadIdx.x;
  const int wave = tid >> 5, lane = tid & 31;
  const int wm = (wave >> 2) * 64;  // 2 wave rows
  const int wn = (wave & 3) * 32;   // 4 wave cols

  floatx8 acc[4][2];
#pragma unroll
  for (int mi = 0; mi < 4; ++mi)
#pragma unroll
    for (int ni = 0; ni < 2; ++ni) {
      floatx8 z = {0.f, 0.f, 0.f, 0.f, 0.f, 0.f, 0.f, 0.f};
      acc[mi][ni] = z;
    }

  for (int k0 = 0; k0 < K_DIM; k0 += TK) {
    // A tile: 128x64 f32 -> bf16, float2 reads + packed b32 LDS stores
#pragma unroll
    for (int i = tid; i < TM * TK / 2; i += 256) {
      int idx = i * 2;
      int r = idx >> 6, c = idx & 63;
      float2 f = *(const float2*)(A + (size_t)(bm + r) * K_DIM + k0 + c);
      *(bf16x2*)&As[r][c] = pack_bf16(f.x, f.y);
    }
    // B tile transposed: coalesced float2 reads along N, b16 scatter into Bt
#pragma unroll
    for (int i = tid; i < TK * TN / 2; i += 256) {
      int idx = i * 2;
      int r = idx >> 7, c = idx & 127;
      float2 f = *(const float2*)(B + (size_t)(k0 + r) * N_QKV + bn + c);
      Bt[c][r]     = (bf16)f.x;
      Bt[c + 1][r] = (bf16)f.y;
    }
    __syncthreads();

#pragma unroll
    for (int kk = 0; kk < TK; kk += 32) {
      bf16x16 afrag[4], bfrag[2];
#pragma unroll
      for (int mi = 0; mi < 4; ++mi)
        afrag[mi] = load_a(&As[0][0], wm + mi * 16, kk, TKP, lane);
#pragma unroll
      for (int ni = 0; ni < 2; ++ni)
        bfrag[ni] = load_b_ct(&Bt[0][0], kk, wn + ni * 16, TKP, lane);
#pragma unroll
      for (int mi = 0; mi < 4; ++mi)
#pragma unroll
        for (int ni = 0; ni < 2; ++ni)
          acc[mi][ni] = wmma_bf16(afrag[mi], bfrag[ni], acc[mi][ni]);
    }
    __syncthreads();
  }

  const int r = lane & 15, h = lane >> 4;
#pragma unroll
  for (int mi = 0; mi < 4; ++mi)
#pragma unroll
    for (int ni = 0; ni < 2; ++ni)
#pragma unroll
      for (int j = 0; j < 8; ++j) {
        int row = bm + wm + mi * 16 + h * 8 + j;
        int col = bn + wn + ni * 16 + r;
        C[(size_t)row * N_QKV + col] = (bf16)acc[mi][ni][j];
      }
}

// ---------------- Attention: LN(q,k) + softmax(QK^T)V per (window, head) ----------------

__global__ __launch_bounds__(128) void attn_kernel(const bf16* __restrict__ qkv,
                                                   const float* __restrict__ gamma,
                                                   const float* __restrict__ beta,
                                                   bf16* __restrict__ attn_out) {
  __shared__ bf16  qs[WINDOW][HEAD_DIM + 8];   // row-major (A operand)
  __shared__ bf16  ks[WINDOW][HEAD_DIM + 8];   // row-major == K^T stored [N][K]
  __shared__ bf16  vt[HEAD_DIM][WINDOW + 8];   // transposed V: vt[d][token]
  __shared__ float Ss[WINDOW][WINDOW + 4];
  __shared__ bf16  ps[WINDOW][WINDOW + 8];

  const int w    = blockIdx.x >> 4;  // window (0..511)
  const int head = blockIdx.x & 15;
  const int tid  = threadIdx.x;
  const int wave = tid >> 5, lane = tid & 31;
  const size_t base = (size_t)w * WINDOW * N_QKV + (size_t)head * HEAD_DIM;

  // LayerNorm: threads 0..63 -> q rows, 64..127 -> k rows. One pass over memory.
  {
    const int row  = tid & 63;
    const int part = tid >> 6;  // 0 = q, 1 = k
    const bf16x8* src8 =
        (const bf16x8*)(qkv + base + (size_t)row * N_QKV + (size_t)part * D_MODEL);
    float f[HEAD_DIM];
    float sum = 0.f, sq = 0.f;
#pragma unroll
    for (int c = 0; c < 8; ++c) {
      bf16x8 chunk = src8[c];
#pragma unroll
      for (int j = 0; j < 8; ++j) {
        float v = (float)chunk[j];
        f[c * 8 + j] = v;
        sum += v;
        sq += v * v;
      }
    }
    float mu   = sum * (1.f / HEAD_DIM);
    float var  = sq * (1.f / HEAD_DIM) - mu * mu;
    float rstd = rsqrtf(var + LN_EPS);
    float smul = (part == 0) ? QK_SCALE : 1.f;
    bf16* dst = (part == 0) ? &qs[row][0] : &ks[row][0];
#pragma unroll
    for (int d = 0; d < HEAD_DIM; d += 2) {
      float v0 = ((f[d]     - mu) * rstd * gamma[d]     + beta[d])     * smul;
      float v1 = ((f[d + 1] - mu) * rstd * gamma[d + 1] + beta[d + 1]) * smul;
      *(bf16x2*)(dst + d) = pack_bf16(v0, v1);
    }
  }
  // Stage V transposed: vt[d][token]  (coalesced global b16 reads along d)
  for (int i = tid; i < WINDOW * HEAD_DIM; i += 128) {
    int d = i & 63, r = i >> 6;
    vt[d][r] = qkv[base + (size_t)r * N_QKV + 2 * D_MODEL + d];
  }
  __syncthreads();

  // S = (Q*scale) @ K^T  (64x64, K=64): 4 waves x 4 N-tiles x 2 K-steps
  {
    const int m0 = wave * 16;
#pragma unroll
    for (int nt = 0; nt < 4; ++nt) {
      floatx8 acc = {0.f, 0.f, 0.f, 0.f, 0.f, 0.f, 0.f, 0.f};
#pragma unroll
      for (int kk = 0; kk < HEAD_DIM; kk += 32) {
        bf16x16 a = load_a(&qs[0][0], m0, kk, HEAD_DIM + 8, lane);
        bf16x16 b = load_b_ct(&ks[0][0], kk, nt * 16, HEAD_DIM + 8, lane);
        acc = wmma_bf16(a, b, acc);
      }
      int r = lane & 15, h = lane >> 4;
#pragma unroll
      for (int j = 0; j < 8; ++j) Ss[m0 + h * 8 + j][nt * 16 + r] = acc[j];
    }
  }
  __syncthreads();

  // Row softmax (threads 0..63, one row each), write P as bf16
  if (tid < WINDOW) {
    float mx = -3.0e38f;
    for (int c = 0; c < WINDOW; ++c) mx = fmaxf(mx, Ss[tid][c]);
    float sum = 0.f;
    for (int c = 0; c < WINDOW; ++c) sum += __expf(Ss[tid][c] - mx);
    float inv = 1.f / sum;
    for (int c = 0; c < WINDOW; c += 2)
      *(bf16x2*)&ps[tid][c] = pack_bf16(__expf(Ss[tid][c] - mx) * inv,
                                        __expf(Ss[tid][c + 1] - mx) * inv);
  }
  __syncthreads();

  // O = P @ V  (64x64, K=64) -> bf16 workspace [token 32768][d_model 1024]
  {
    const int m0 = wave * 16;
#pragma unroll
    for (int nt = 0; nt < 4; ++nt) {
      floatx8 acc = {0.f, 0.f, 0.f, 0.f, 0.f, 0.f, 0.f, 0.f};
#pragma unroll
      for (int kk = 0; kk < WINDOW; kk += 32) {
        bf16x16 a = load_a(&ps[0][0], m0, kk, WINDOW + 8, lane);
        bf16x16 b = load_b_ct(&vt[0][0], kk, nt * 16, WINDOW + 8, lane);
        acc = wmma_bf16(a, b, acc);
      }
      int r = lane & 15, h = lane >> 4;
#pragma unroll
      for (int j = 0; j < 8; ++j) {
        int row = m0 + h * 8 + j;
        int col = nt * 16 + r;
        attn_out[((size_t)w * WINDOW + row) * D_MODEL + (size_t)head * HEAD_DIM + col] =
            (bf16)acc[j];
      }
    }
  }
}

// ---------------- GEMM 2: out = attn(bf16) @ W_proj(f32) + b_proj -> f32 ----------------

__global__ __launch_bounds__(256) void gemm_proj_kernel(const bf16* __restrict__ A,
                                                        const float* __restrict__ B,
                                                        const float* __restrict__ bias,
                                                        float* __restrict__ C) {
  __shared__ bf16 As[TM][TKP];
  __shared__ bf16 Bt[TN][TKP];
  const int bm = blockIdx.x * TM;
  const int bn = blockIdx.y * TN;
  const int tid = threadIdx.x;
  const int wave = tid >> 5, lane = tid & 31;
  const int wm = (wave >> 2) * 64;
  const int wn = (wave & 3) * 32;
  const unsigned int as_base = (unsigned int)(size_t)&As[0][0];

  floatx8 acc[4][2];
#pragma unroll
  for (int mi = 0; mi < 4; ++mi)
#pragma unroll
    for (int ni = 0; ni < 2; ++ni) {
      floatx8 z = {0.f, 0.f, 0.f, 0.f, 0.f, 0.f, 0.f, 0.f};
      acc[mi][ni] = z;
    }

  for (int k0 = 0; k0 < K_DIM; k0 += TK) {
    // A tile is already bf16 & row-major: async DMA global->LDS, 16B chunks.
    // 128 rows x 8 chunks = 1024 chunks, 4 per thread.
#pragma unroll
    for (int j = tid; j < TM * (TK / 8); j += 256) {
      int r = j >> 3, cc = j & 7;
      unsigned int lds_off = as_base + (unsigned int)(r * TKP + cc * 8) * 2u;
      async_load_b128(lds_off, A + (size_t)(bm + r) * K_DIM + k0 + cc * 8);
    }
    // W_proj tile transposed with conversion
#pragma unroll
    for (int i = tid; i < TK * TN / 2; i += 256) {
      int idx = i * 2;
      int r = idx >> 7, c = idx & 127;
      float2 f = *(const float2*)(B + (size_t)(k0 + r) * D_MODEL + bn + c);
      Bt[c][r]     = (bf16)f.x;
      Bt[c + 1][r] = (bf16)f.y;
    }
    wait_asynccnt0();
    __syncthreads();

#pragma unroll
    for (int kk = 0; kk < TK; kk += 32) {
      bf16x16 afrag[4], bfrag[2];
#pragma unroll
      for (int mi = 0; mi < 4; ++mi)
        afrag[mi] = load_a(&As[0][0], wm + mi * 16, kk, TKP, lane);
#pragma unroll
      for (int ni = 0; ni < 2; ++ni)
        bfrag[ni] = load_b_ct(&Bt[0][0], kk, wn + ni * 16, TKP, lane);
#pragma unroll
      for (int mi = 0; mi < 4; ++mi)
#pragma unroll
        for (int ni = 0; ni < 2; ++ni)
          acc[mi][ni] = wmma_bf16(afrag[mi], bfrag[ni], acc[mi][ni]);
    }
    __syncthreads();
  }

  const int r = lane & 15, h = lane >> 4;
#pragma unroll
  for (int ni = 0; ni < 2; ++ni) {
    float bv = bias[bn + wn + ni * 16 + r];
#pragma unroll
    for (int mi = 0; mi < 4; ++mi)
#pragma unroll
      for (int j = 0; j < 8; ++j) {
        int row = bm + wm + mi * 16 + h * 8 + j;
        int col = bn + wn + ni * 16 + r;
        C[(size_t)row * D_MODEL + col] = acc[mi][ni][j] + bv;
      }
  }
}

// ---------------- host ----------------

extern "C" void kernel_launch(void* const* d_in, const int* in_sizes, int n_in,
                              void* d_out, int out_size, void* d_ws, size_t ws_size,
                              hipStream_t stream) {
  const float* x        = (const float*)d_in[0];
  const float* W_qkv    = (const float*)d_in[1];
  const float* ln_gamma = (const float*)d_in[2];
  const float* ln_beta  = (const float*)d_in[3];
  const float* W_proj   = (const float*)d_in[4];
  const float* b_proj   = (const float*)d_in[5];
  float* out = (float*)d_out;

  // Workspace: qkv bf16 (32768 x 3072 = 192 MiB) + attn bf16 (32768 x 1024 = 64 MiB)
  bf16* qkv_ws  = (bf16*)d_ws;
  bf16* attn_ws = qkv_ws + (size_t)M_TOT * N_QKV;

  dim3 g1(M_TOT / TM, N_QKV / TN);
  gemm_qkv_kernel<<<g1, 256, 0, stream>>>(x, W_qkv, qkv_ws);

  attn_kernel<<<512 * NUM_HEADS, 128, 0, stream>>>(qkv_ws, ln_gamma, ln_beta, attn_ws);

  dim3 g2(M_TOT / TM, D_MODEL / TN);
  gemm_proj_kernel<<<g2, 256, 0, stream>>>(attn_ws, W_proj, b_proj, out);
}
